// MultiHeadAttentionWithRoPE_57775900066380
// MI455X (gfx1250) — matrix operand order repaired and checked by reading, not compiled, not run on hardware
//
#include <hip/hip_runtime.h>

// ---------------------------------------------------------------------------
// MHA with RoPE for MI455X (gfx1250, wave32, WMMA bf16 + async-to-LDS path)
// B=4, S=2048, D=1024, H=16, hd=64
// ---------------------------------------------------------------------------

#define B_   4
#define S_   2048
#define D_   1024
#define H_   16
#define HD_  64
#define N3_  3072   // 3*D

typedef __attribute__((ext_vector_type(16))) __bf16 v16bf;
typedef __attribute__((ext_vector_type(8)))  float  v8f;

union Frag { v16bf v; unsigned u[8]; };

__device__ inline v8f wmma_bf16(v16bf a, v16bf b, v8f c) {
  return __builtin_amdgcn_wmma_f32_16x16x32_bf16(
      /*neg_a=*/false, a, /*neg_b=*/false, b,
      /*c_mod=*/(short)0, c, /*reuse_a=*/false, /*reuse_b=*/false);
}

__device__ inline v8f vzero8() {
  v8f z;
  #pragma unroll
  for (int i = 0; i < 8; ++i) z[i] = 0.0f;
  return z;
}

__device__ inline unsigned short f2bf(float f) {
  union { float f; unsigned u; } x; x.f = f;
  unsigned r = x.u + 0x7FFFu + ((x.u >> 16) & 1u);   // round-to-nearest-even
  return (unsigned short)(r >> 16);
}
__device__ inline float bf2f(unsigned short h) {
  union { unsigned u; float f; } x; x.u = ((unsigned)h) << 16;
  return x.f;
}

// Issue one 16-byte async global->LDS copy (ASYNCcnt-tracked, GV mode).
// ldsa = workgroup-relative LDS byte address (low 32 bits of flat shared ptr),
// gptr = 16B-aligned global address.
__device__ inline void async_copy_b128(const void* gptr, void* lds_ptr) {
  unsigned ldsa = (unsigned)(size_t)lds_ptr;            // addr[31:0] = LDS offset
  unsigned long long ga = (unsigned long long)(size_t)gptr;
  asm volatile("global_load_async_to_lds_b128 %0, %1, off"
               :: "v"(ldsa), "v"(ga) : "memory");
}
__device__ inline void wait_async0() {
  asm volatile("s_wait_asynccnt 0" ::: "memory");
}

// ---------------------------------------------------------------------------
// fp32 -> bf16 elementwise convert
// ---------------------------------------------------------------------------
__global__ __launch_bounds__(256) void k_convert_bf16(
    const float* __restrict__ in, unsigned short* __restrict__ out, int n) {
  int i = blockIdx.x * 256 + threadIdx.x;
  if (i < n) out[i] = f2bf(in[i]);
}

// fp32 [K,N] -> bf16 transposed [N,K] (so GEMM B-tile rows are contiguous K)
__global__ __launch_bounds__(256) void k_transpose_bf16(
    const float* __restrict__ in, unsigned short* __restrict__ out, int Kd, int N) {
  int i = blockIdx.x * 256 + threadIdx.x;
  if (i >= Kd * N) return;
  int k = i / N, n = i - k * N;
  out[(size_t)n * Kd + k] = f2bf(in[i]);
}

// ---------------------------------------------------------------------------
// bf16 WMMA GEMM: C[M,N] = A[M,K] @ B[K,N] + bias, B given transposed [N,K].
// 256 threads = 8 waves. Block tile 128M x 64N; per K-step of 32 the A tile
// (128x32) and B tile (64x32) are staged in LDS via async-to-LDS b128 copies
// (coalesced 16B transfers, B shared by all 8 waves), then consumed as WMMA
// fragments with LDS dword loads. Wave w owns rows [wave*16, wave*16+16).
// ---------------------------------------------------------------------------
template <bool OUT_BF16>
__global__ __launch_bounds__(256) void k_gemm_bf16_wmma(
    const unsigned short* __restrict__ A,   // bf16 [M,K]
    const unsigned short* __restrict__ BT,  // bf16 [N,K]
    const float* __restrict__ bias,         // [N]
    void* __restrict__ Cout, int M, int N, int Kd) {
  __shared__ unsigned short at[128 * 32];   // [mrow_local][k_local]  8 KB
  __shared__ unsigned short bt[64 * 32];    // [ncol_local][k_local]  4 KB

  const int tid = threadIdx.x;
  const int lane = tid & 31;
  const int wave = tid >> 5;
  const int lo = lane & 15, hi = lane >> 4;
  const int mBase = blockIdx.y * 128;
  const int nBase = blockIdx.x * 64;

  const unsigned* at_dw = (const unsigned*)at;
  const unsigned* bt_dw = (const unsigned*)bt;

  v8f acc[4];
  #pragma unroll
  for (int i = 0; i < 4; ++i) acc[i] = vzero8();

  for (int kk = 0; kk < Kd; kk += 32) {
    __syncthreads();
    // --- stage A tile: 128 rows x 32 K = 512 x b128, 2 per thread ---
    #pragma unroll
    for (int i = 0; i < 2; ++i) {
      int t = tid + i * 256;                // 0..511
      int r = t >> 2, ch = t & 3;           // row, 8-element chunk
      async_copy_b128(A + (size_t)(mBase + r) * Kd + kk + ch * 8,
                      &at[r * 32 + ch * 8]);
    }
    // --- stage B tile: 64 cols x 32 K = 256 x b128, 1 per thread ---
    {
      int r = tid >> 2, ch = tid & 3;
      async_copy_b128(BT + (size_t)(nBase + r) * Kd + kk + ch * 8,
                      &bt[r * 32 + ch * 8]);
    }
    wait_async0();
    __syncthreads();

    Frag a;
    #pragma unroll
    for (int v = 0; v < 8; ++v) {
      int koff = (v < 4 ? 2 * v : 8 + 2 * v) + 8 * hi;   // A-frag K layout
      a.u[v] = at_dw[(wave * 16 + lo) * 16 + (koff >> 1)];
    }
    #pragma unroll
    for (int nt = 0; nt < 4; ++nt) {
      Frag b;
      #pragma unroll
      for (int v = 0; v < 8; ++v)                        // B-frag: same K layout as A-frag
        b.u[v] = bt_dw[(nt * 16 + lo) * 16 + (v < 4 ? v : v + 4) + 4 * hi];
      acc[nt] = wmma_bf16(a.v, b.v, acc[nt]);
    }
  }

  #pragma unroll
  for (int nt = 0; nt < 4; ++nt) {
    #pragma unroll
    for (int r = 0; r < 8; ++r) {
      int row = mBase + wave * 16 + r + 8 * hi;          // C layout: M = r + 8*hi
      int col = nBase + nt * 16 + lo;
      float val = acc[nt][r] + bias[col];
      if (OUT_BF16)
        ((unsigned short*)Cout)[(size_t)row * N + col] = f2bf(val);
      else
        ((float*)Cout)[(size_t)row * N + col] = val;
    }
  }
}

// ---------------------------------------------------------------------------
// RoPE + head split: qkv bf16 [B,S,3D] -> Q,K,V bf16 [B,H,S,hd]
// one thread per (b,s,h,j), j in [0,32): handles rotation pair (j, j+32)
// ---------------------------------------------------------------------------
__global__ __launch_bounds__(256) void k_rope_split(
    const unsigned short* __restrict__ qkv,
    unsigned short* __restrict__ Qb, unsigned short* __restrict__ Kb,
    unsigned short* __restrict__ Vb) {
  int idx = blockIdx.x * 256 + threadIdx.x;   // 4*2048*16*32 = 2^22 threads exact
  int j = idx & 31;
  int h = (idx >> 5) & 15;
  int s = (idx >> 9) & 2047;
  int b = idx >> 20;

  // inv_freq[j] = 10000^(-2j/64) = exp(-j * ln(10000)/32)
  float invfreq = __expf(-(float)j * (9.210340371976184f / 32.0f));
  float ang = (float)s * invfreq;
  float sn, c;
  __sincosf(ang, &sn, &c);

  size_t base = ((size_t)(b * S_ + s)) * N3_ + h * HD_ + j;
  float q1 = bf2f(qkv[base]);
  float q2 = bf2f(qkv[base + 32]);
  float k1 = bf2f(qkv[base + D_]);
  float k2 = bf2f(qkv[base + D_ + 32]);

  size_t dst = ((size_t)((b * H_ + h) * S_ + s)) * HD_ + j;
  Qb[dst]      = f2bf(q1 * c - q2 * sn);
  Qb[dst + 32] = f2bf(q2 * c + q1 * sn);
  Kb[dst]      = f2bf(k1 * c - k2 * sn);
  Kb[dst + 32] = f2bf(k2 * c + k1 * sn);
  Vb[dst]      = qkv[base + 2 * D_];
  Vb[dst + 32] = qkv[base + 2 * D_ + 32];
}

// ---------------------------------------------------------------------------
// Flash attention (bf16 WMMA, fp32 online softmax).
// grid = (S/128, H, B); block = 256 (8 waves). Wave w owns 16 query rows.
// K-tile [64 key][64 d] staged via async-to-LDS b128; V^T-tile [64 d][64 key]
// staged with a manual transpose (async path cannot transpose). Both shared
// by all 8 waves. Output written bf16 in [B,S,D] layout for the final GEMM.
// ---------------------------------------------------------------------------
__global__ __launch_bounds__(256) void k_flash_attn(
    const unsigned short* __restrict__ Q,   // [B,H,S,hd] bf16
    const unsigned short* __restrict__ K,   // [B,H,S,hd] bf16
    const unsigned short* __restrict__ V,   // [B,H,S,hd] bf16
    unsigned short* __restrict__ O) {       // [B,S,D] bf16
  __shared__ unsigned kt[64 * 32];          // [key][d] bf16 pairs as dwords
  __shared__ unsigned vt[64 * 32];          // [d][key] bf16 pairs as dwords
  __shared__ unsigned short pbuf[8][16 * 64];  // per-wave P tile, row-major

  const int tid = threadIdx.x;
  const int lane = tid & 31, wave = tid >> 5;
  const int lo = lane & 15, hi = lane >> 4;
  const int qt = blockIdx.x, h = blockIdx.y, b = blockIdx.z;
  const int bh = b * H_ + h;
  const int qBase = qt * 128 + wave * 16;

  const unsigned* Qd = (const unsigned*)Q;

  // Q fragments: 16 rows x 64 d = two 16x32 A-fragments, kept in registers.
  Frag qf[2];
  {
    const int row = qBase + lo;
    const size_t base = ((size_t)bh * S_ + row) * 32;   // dwords per row = 32
    #pragma unroll
    for (int c = 0; c < 2; ++c)
      #pragma unroll
      for (int v = 0; v < 8; ++v) {
        int koff = (v < 4 ? 2 * v : 8 + 2 * v) + 8 * hi + c * 32;
        qf[c].u[v] = Qd[base + (koff >> 1)];
      }
  }

  float m8[8], l8[8];
  v8f o4[4];
  #pragma unroll
  for (int r = 0; r < 8; ++r) { m8[r] = -1e30f; l8[r] = 0.0f; }
  #pragma unroll
  for (int i = 0; i < 4; ++i) o4[i] = vzero8();

  for (int kb = 0; kb < S_ / 64; ++kb) {
    __syncthreads();
    // --- stage K tile via async-to-LDS: 8 KB = 512 x b128, 2 per thread ---
    {
      const unsigned short* kp = K + ((size_t)bh * S_ + kb * 64) * HD_;
      #pragma unroll
      for (int i = 0; i < 2; ++i) {
        int t = tid + i * 256;                 // 0..511 : b128 chunk id
        async_copy_b128(kp + t * 8, &kt[t * 4]);
      }
      // --- stage V^T tile (element transpose, 16 elems per thread) ---
      unsigned short* vts = (unsigned short*)vt;
      #pragma unroll
      for (int i = 0; i < 16; ++i) {
        int e = tid * 16 + i;
        int key = e >> 6, d = e & 63;
        vts[d * 64 + key] = V[((size_t)bh * S_ + kb * 64 + key) * HD_ + d];
      }
    }
    wait_async0();
    __syncthreads();

    // --- scores: S16x64 = Q(16x64) @ K^T, 8 WMMA ---
    v8f sc[4];
    #pragma unroll
    for (int nt = 0; nt < 4; ++nt) {
      sc[nt] = vzero8();
      #pragma unroll
      for (int c = 0; c < 2; ++c) {
        Frag bk;
        #pragma unroll
        for (int v = 0; v < 8; ++v) {
          int key = nt * 16 + lo;
          int d = c * 32 + (v < 4 ? 2 * v : 8 + 2 * v) + 8 * hi;
          bk.u[v] = kt[key * 32 + (d >> 1)];
        }
        sc[nt] = wmma_bf16(qf[c].v, bk.v, sc[nt]);
      }
      #pragma unroll
      for (int r = 0; r < 8; ++r) sc[nt][r] *= 0.125f;  // 1/sqrt(64)
    }

    // --- online softmax: row stats live in 16-lane groups (C-layout N stripe)
    float mn[8], corr[8];
    #pragma unroll
    for (int r = 0; r < 8; ++r) {
      float tm = fmaxf(fmaxf(sc[0][r], sc[1][r]), fmaxf(sc[2][r], sc[3][r]));
      #pragma unroll
      for (int mask = 1; mask < 16; mask <<= 1)
        tm = fmaxf(tm, __shfl_xor(tm, mask));
      mn[r] = fmaxf(m8[r], tm);
      corr[r] = __expf(m8[r] - mn[r]);
    }
    #pragma unroll
    for (int nt = 0; nt < 4; ++nt)
      #pragma unroll
      for (int r = 0; r < 8; ++r)
        sc[nt][r] = __expf(sc[nt][r] - mn[r]);
    #pragma unroll
    for (int r = 0; r < 8; ++r) {
      float s = sc[0][r] + sc[1][r] + sc[2][r] + sc[3][r];
      #pragma unroll
      for (int mask = 1; mask < 16; mask <<= 1) s += __shfl_xor(s, mask);
      l8[r] = l8[r] * corr[r] + s;
      m8[r] = mn[r];
    }
    #pragma unroll
    for (int nt = 0; nt < 4; ++nt)
      #pragma unroll
      for (int r = 0; r < 8; ++r) o4[nt][r] *= corr[r];

    // --- P (C-layout) -> LDS -> A-fragment layout ---
    #pragma unroll
    for (int nt = 0; nt < 4; ++nt)
      #pragma unroll
      for (int r = 0; r < 8; ++r)
        pbuf[wave][(r + 8 * hi) * 64 + nt * 16 + lo] = f2bf(sc[nt][r]);
    asm volatile("s_wait_dscnt 0" ::: "memory");  // wave-private LDS RAW

    const unsigned* pb = (const unsigned*)pbuf[wave];
    Frag pa[2];
    #pragma unroll
    for (int c = 0; c < 2; ++c)
      #pragma unroll
      for (int v = 0; v < 8; ++v) {
        int koff = (v < 4 ? 2 * v : 8 + 2 * v) + 8 * hi + c * 32;
        pa[c].u[v] = pb[lo * 32 + (koff >> 1)];
      }

    // --- O += P @ V, 8 WMMA ---
    #pragma unroll
    for (int nt = 0; nt < 4; ++nt)
      #pragma unroll
      for (int c = 0; c < 2; ++c) {
        Frag bv;
        #pragma unroll
        for (int v = 0; v < 8; ++v) {
          int key = c * 32 + (v < 4 ? 2 * v : 8 + 2 * v) + 8 * hi;
          int d = nt * 16 + lo;
          bv.u[v] = vt[d * 32 + (key >> 1)];
        }
        o4[nt] = wmma_bf16(pa[c].v, bv.v, o4[nt]);
      }
  }

  // --- epilogue: normalize and store bf16 in [B, S, H*hd] layout ---
  #pragma unroll
  for (int nt = 0; nt < 4; ++nt)
    #pragma unroll
    for (int r = 0; r < 8; ++r) {
      float val = o4[nt][r] / l8[r];
      int row = qBase + r + 8 * hi;
      int col = h * HD_ + nt * 16 + lo;
      O[((size_t)b * S_ + row) * D_ + col] = f2bf(val);
    }
}

// ---------------------------------------------------------------------------
// Host launch
// ---------------------------------------------------------------------------
extern "C" void kernel_launch(void* const* d_in, const int* in_sizes, int n_in,
                              void* d_out, int out_size, void* d_ws, size_t ws_size,
                              hipStream_t stream) {
  (void)in_sizes; (void)n_in; (void)out_size; (void)ws_size;
  const float* x      = (const float*)d_in[0];  // [B,S,D]
  const float* w_qkv  = (const float*)d_in[1];  // [D,3D]
  const float* b_qkv  = (const float*)d_in[2];  // [3D]
  const float* w_out  = (const float*)d_in[3];  // [D,D]
  const float* b_out  = (const float*)d_in[4];  // [D]
  float* out          = (float*)d_out;          // [B,S,D]

  const size_t M = (size_t)B_ * S_;             // 8192
  char* ws = (char*)d_ws;
  unsigned short* Xbf   = (unsigned short*)(ws);                       // 16 MiB
  unsigned short* WqkvT = (unsigned short*)(ws + 16777216);            //  6 MiB
  unsigned short* WoutT = (unsigned short*)(ws + 23068672);            //  2 MiB
  unsigned short* QKVbf = (unsigned short*)(ws + 25165824);            // 48 MiB
  unsigned short* Qbf   = (unsigned short*)(ws + 75497472);            // 16 MiB
  unsigned short* Kbf   = (unsigned short*)(ws + 92274688);            // 16 MiB
  unsigned short* Vbf   = (unsigned short*)(ws + 109051904);           // 16 MiB
  unsigned short* Abf   = (unsigned short*)(ws + 125829120);           // 16 MiB

  // 1) converts
  {
    int n = (int)(M * D_);
    k_convert_bf16<<<(n + 255) / 256, 256, 0, stream>>>(x, Xbf, n);
    int nq = D_ * N3_;
    k_transpose_bf16<<<(nq + 255) / 256, 256, 0, stream>>>(w_qkv, WqkvT, D_, N3_);
    int no = D_ * D_;
    k_transpose_bf16<<<(no + 255) / 256, 256, 0, stream>>>(w_out, WoutT, D_, D_);
  }

  // 2) QKV GEMM (bf16 out, bias fused): [8192,1024] @ [1024,3072]
  {
    dim3 grid(N3_ / 64, (int)(M / 128));
    k_gemm_bf16_wmma<true><<<grid, 256, 0, stream>>>(
        Xbf, WqkvT, b_qkv, QKVbf, (int)M, N3_, D_);
  }

  // 3) RoPE + head split
  {
    int n = B_ * S_ * H_ * (HD_ / 2);   // 2^22
    k_rope_split<<<n / 256, 256, 0, stream>>>(QKVbf, Qbf, Kbf, Vbf);
  }

  // 4) flash attention
  {
    dim3 grid(S_ / 128, H_, B_);
    k_flash_attn<<<grid, 256, 0, stream>>>(Qbf, Kbf, Vbf, Abf);
  }

  // 5) output projection (fp32 out, bias fused): [8192,1024] @ [1024,1024]
  {
    dim3 grid(D_ / 64, (int)(M / 128));
    k_gemm_bf16_wmma<false><<<grid, 256, 0, stream>>>(
        Abf, WoutT, b_out, out, (int)M, D_, D_);
  }
}
